// MAE_12902081757221
// MI455X (gfx1250) — matrix operand order repaired and checked by reading, not compiled, not run on hardware
//
#include <hip/hip_runtime.h>
#include <hip/hip_bf16.h>
#include <math.h>

// ---- CDNA5 WMMA vector types ----
typedef __attribute__((ext_vector_type(16))) __bf16 v16bf;
typedef __attribute__((ext_vector_type(8)))  float  v8f;
typedef __attribute__((ext_vector_type(4)))  int    v4i;

#define MTOK   3136   // 64 batches * 49 kept tokens
#define DIMC   768
#define NHEAD  12
#define DHEAD  64
#define FFD    3072
#define KEEPT  49

#define AS1 __attribute__((address_space(1)))
#define AS3 __attribute__((address_space(3)))

#if defined(__HIP_DEVICE_COMPILE__) && __has_builtin(__builtin_amdgcn_global_load_async_to_lds_b128)
#define HAVE_ASYNC_LDS 1
#else
#define HAVE_ASYNC_LDS 0
#endif

static __device__ __forceinline__ __bf16 f2bf(float x) { return (__bf16)x; }
static __device__ __forceinline__ float  bf2f(__bf16 x){ return (float)x; }

// pack two f32 -> two bf16 in one dword (low address = low half)
static __device__ __forceinline__ unsigned pack2bf(float a, float b) {
  union { unsigned u; __bf16 h[2]; } p;
  p.h[0] = f2bf(a); p.h[1] = f2bf(b);
  return p.u;
}

#if HAVE_ASYNC_LDS
static __device__ __forceinline__ void wait_async0() {
#if __has_builtin(__builtin_amdgcn_s_wait_asynccnt)
  __builtin_amdgcn_s_wait_asynccnt(0);
#else
  asm volatile("s_wait_asynccnt 0x0" ::: "memory");
#endif
}
#endif

// branchless tanh-GELU using the v_exp_f32 TRANS unit
static __device__ __forceinline__ float gelu_tanh(float x) {
  float u = 0.7978845608028654f * (x + 0.044715f * x * x * x);
  u = fminf(fmaxf(u, -10.0f), 10.0f);           // avoid inf/inf
  const float e = __expf(2.0f * u);
  const float t = (e - 1.0f) * __builtin_amdgcn_rcpf(e + 1.0f);
  return 0.5f * x * (1.0f + t);
}

// =====================================================================
// bf16-WMMA GEMM:  out = act(A[M,K](bf16) @ W[K,N](f32->bf16) [+ bias])
// Compile-time epilogue: GELU, fp32 residual accumulate, bf16 store.
// Tile: 64(M) x 128(N) per 128-thread workgroup, BK=32.
// 4 waves in a 2x2 grid, each computing 32(M) x 64(N)  -> 8 WMMA / K-step.
// M = gridDim.y*64 (3136), N % 128 == 0, K % 32 == 0.
// =====================================================================
template <bool HASBIAS, bool GELU, bool RESID, bool STOREBF>
__global__ __launch_bounds__(128) void gemm_wmma(
    const __bf16* __restrict__ A, const float* __restrict__ W,
    const float* __restrict__ bias, float* __restrict__ Xres,
    __bf16* __restrict__ Obf, int N, int K)
{
  __shared__ __bf16 sA[64][32];    // [m][k]
  __shared__ __bf16 sBT[128][32];  // [n][k]  (transposed so K is contiguous)

  const int tid  = threadIdx.x;
  const int lane = tid & 31;
  const int wave = tid >> 5;
  const int wm   = (wave >> 1) * 32;   // 0 / 32
  const int wn   = (wave & 1) * 64;    // 0 / 64
  const int bm   = blockIdx.y * 64;
  const int bn   = blockIdx.x * 128;

  const int ln  = lane & 15;
  const int h8  = (lane >> 4) * 8;
  const int h16 = (lane >> 4) * 16;
  const int hi8 = (lane >> 4) * 8;

  const int lA_r = tid >> 1;           // 0..63
  const int lA_c = (tid & 1) * 16;     // 0 or 16
  const int lBk2 = tid >> 3;           // k-pair index 0..15
  const int lBn  = (tid & 7) * 16;     // n chunk base 0..112

  const v8f zero = {0.f,0.f,0.f,0.f,0.f,0.f,0.f,0.f};
  v8f acc[2][4];
#pragma unroll
  for (int i = 0; i < 2; ++i)
#pragma unroll
    for (int j = 0; j < 4; ++j) acc[i][j] = zero;

  for (int k0 = 0; k0 < K; k0 += 32) {
    // ---- stage A tile (bf16 copy): async DMA to LDS when available ----
#if HAVE_ASYNC_LDS
    {
      const __bf16* gsrc = A + (size_t)(bm + lA_r) * K + k0 + lA_c;
      __builtin_amdgcn_global_load_async_to_lds_b128(
          (AS1 v4i*)gsrc, (AS3 v4i*)&sA[lA_r][lA_c], 0, 0);
      __builtin_amdgcn_global_load_async_to_lds_b128(
          (AS1 v4i*)gsrc, (AS3 v4i*)&sA[lA_r][lA_c], 16, 0);
    }
#else
    {
      const __bf16* ap = A + (size_t)(bm + lA_r) * K + k0 + lA_c;
#pragma unroll
      for (int i = 0; i < 16; ++i) sA[lA_r][lA_c + i] = ap[i];
    }
#endif
    // ---- stage W tile: coalesced f32 reads of a K row-pair, packed
    //      dword stores into the transposed LDS tile ----
    {
      const float* r0 = W + (size_t)(k0 + 2 * lBk2) * N + bn + lBn;
      const float* r1 = r0 + N;
#pragma unroll
      for (int i = 0; i < 16; ++i)
        *(unsigned*)&sBT[lBn + i][lBk2 * 2] = pack2bf(r0[i], r1[i]);
    }
#if HAVE_ASYNC_LDS
    wait_async0();
#endif
    __syncthreads();

    v16bf af[2], bfg[4];
#pragma unroll
    for (int t = 0; t < 2; ++t) {
      const int m = wm + t * 16 + ln;
#pragma unroll
      for (int e = 0; e < 8; ++e) {
        af[t][e]     = sA[m][h16 + e];       // K = h16+e
        af[t][8 + e] = sA[m][h16 + 8 + e];   // K = h16+8+e
      }
    }
#pragma unroll
    for (int j = 0; j < 4; ++j) {
      const int n = wn + j * 16 + ln;
#pragma unroll
      for (int e = 0; e < 16; ++e) bfg[j][e] = sBT[n][h16 + e];
    }
#pragma unroll
    for (int i = 0; i < 2; ++i)
#pragma unroll
      for (int j = 0; j < 4; ++j)
        acc[i][j] = __builtin_amdgcn_wmma_f32_16x16x32_bf16(
            false, af[i], false, bfg[j], (short)0, acc[i][j], false, false);
    __syncthreads();
  }

  // ---- straight-line epilogue (compile-time variants) ----
#pragma unroll
  for (int i = 0; i < 2; ++i) {
#pragma unroll
    for (int j = 0; j < 4; ++j) {
      const int ncol = bn + wn + j * 16 + ln;
      float bv = 0.0f;
      if (HASBIAS) bv = bias[ncol];
#pragma unroll
      for (int r = 0; r < 8; ++r) {
        const int mrow = bm + wm + i * 16 + r + hi8;
        float v = acc[i][j][r] + bv;
        if (GELU) v = gelu_tanh(v);
        if (RESID)   Xres[(size_t)mrow * N + ncol] += v;
        if (STOREBF) Obf[(size_t)mrow * N + ncol] = f2bf(v);
      }
    }
  }
}

// =====================================================================
// Patch embed: gathered GEMM over only the 49 selected patches per image
// (mask_idx >= 1 always -> the class token is never selected; index 197
// clamps to 196 per JAX OOB semantics).
// Out: X[m,n] = patch(m) . conv_w[:,n] + pe[idx(m), n]   (fp32).
// The (ph,pw,c) address decomposition is precomputed once per block into
// an LDS offset table to avoid per-element integer division.
// =====================================================================
__global__ __launch_bounds__(128) void patch_embed(
    const float* __restrict__ img, const float* __restrict__ convw,
    const float* __restrict__ pe, const int* __restrict__ midx,
    float* __restrict__ X)
{
  const int N = DIMC, K = DIMC;
  __shared__ __bf16 sA[64][32];
  __shared__ __bf16 sBT[128][32];
  __shared__ int    offs[DIMC];    // k -> (ph*224+pw)*3+c

  const int tid  = threadIdx.x;
  const int lane = tid & 31;
  const int wave = tid >> 5;
  const int wm   = (wave >> 1) * 32;
  const int wn   = (wave & 1) * 64;
  const int bm   = blockIdx.y * 64;
  const int bn   = blockIdx.x * 128;

  const int ln  = lane & 15;
  const int h8  = (lane >> 4) * 8;
  const int h16 = (lane >> 4) * 16;
  const int hi8 = (lane >> 4) * 8;

  const int lA_r = tid >> 1;
  const int lA_c = (tid & 1) * 16;
  const int lBk2 = tid >> 3;
  const int lBn  = (tid & 7) * 16;

  // build gather-offset table (once per block)
  for (int k = tid; k < DIMC; k += 128) {
    const int ph = k / 48;
    const int rr = k - ph * 48;
    const int pw = rr / 3;
    const int c  = rr - pw * 3;
    offs[k] = (ph * 224 + pw) * 3 + c;
  }

  // per-thread gather base for its A row
  const int m_r = bm + lA_r;
  const int b_img = m_r / KEEPT;
  const int j_tok = m_r % KEEPT;
  int idx = midx[j_tok]; if (idx > 196) idx = 196;
  const int p  = idx - 1;
  const int py = p / 14, px = p % 14;
  const float* ibase = img + ((size_t)(b_img * 224 + py * 16) * 224 + px * 16) * 3;

  const v8f zero = {0.f,0.f,0.f,0.f,0.f,0.f,0.f,0.f};
  v8f acc[2][4];
#pragma unroll
  for (int i = 0; i < 2; ++i)
#pragma unroll
    for (int j = 0; j < 4; ++j) acc[i][j] = zero;

  __syncthreads();   // offs table ready

  for (int k0 = 0; k0 < K; k0 += 32) {
#pragma unroll
    for (int i = 0; i < 16; ++i)
      sA[lA_r][lA_c + i] = f2bf(ibase[offs[k0 + lA_c + i]]);
    {
      const float* r0 = convw + (size_t)(k0 + 2 * lBk2) * N + bn + lBn;
      const float* r1 = r0 + N;
#pragma unroll
      for (int i = 0; i < 16; ++i)
        *(unsigned*)&sBT[lBn + i][lBk2 * 2] = pack2bf(r0[i], r1[i]);
    }
    __syncthreads();

    v16bf af[2], bfg[4];
#pragma unroll
    for (int t = 0; t < 2; ++t) {
      const int m = wm + t * 16 + ln;
#pragma unroll
      for (int e = 0; e < 8; ++e) {
        af[t][e]     = sA[m][h16 + e];
        af[t][8 + e] = sA[m][h16 + 8 + e];
      }
    }
#pragma unroll
    for (int j = 0; j < 4; ++j) {
      const int n = wn + j * 16 + ln;
#pragma unroll
      for (int e = 0; e < 16; ++e) bfg[j][e] = sBT[n][h16 + e];
    }
#pragma unroll
    for (int i = 0; i < 2; ++i)
#pragma unroll
      for (int j = 0; j < 4; ++j)
        acc[i][j] = __builtin_amdgcn_wmma_f32_16x16x32_bf16(
            false, af[i], false, bfg[j], (short)0, acc[i][j], false, false);
    __syncthreads();
  }

  // epilogue: incremental %49, pe row pointers hoisted out of the j-loop
#pragma unroll
  for (int i = 0; i < 2; ++i) {
    const int mbase = bm + wm + i * 16 + hi8;
    const int j0 = mbase % KEEPT;
    const float* peRow[8];
#pragma unroll
    for (int r = 0; r < 8; ++r) {
      int jt = j0 + r;
      jt = (jt >= KEEPT) ? jt - KEEPT : jt;
      int pidx = midx[jt]; if (pidx > 196) pidx = 196;
      peRow[r] = pe + (size_t)pidx * DIMC;
    }
#pragma unroll
    for (int j = 0; j < 4; ++j) {
      const int ncol = bn + wn + j * 16 + ln;
#pragma unroll
      for (int r = 0; r < 8; ++r)
        X[(size_t)(mbase + r) * N + ncol] = acc[i][j][r] + peRow[r][ncol];
    }
  }
}

// =====================================================================
// LayerNorm over DIM=768, one token per 256-thread block; bf16 output.
// =====================================================================
__global__ __launch_bounds__(256) void layernorm(
    const float* __restrict__ X, const float* __restrict__ gs,
    const float* __restrict__ gb, __bf16* __restrict__ H)
{
  const int m = blockIdx.x;
  const int t = threadIdx.x;
  const float* x = X + (size_t)m * DIMC;
  __shared__ float red[256];

  float v0 = x[t], v1 = x[t + 256], v2 = x[t + 512];
  red[t] = v0 + v1 + v2;
  __syncthreads();
  for (int off = 128; off > 0; off >>= 1) {
    if (t < off) red[t] += red[t + off];
    __syncthreads();
  }
  const float mean = red[0] * (1.0f / 768.0f);
  __syncthreads();
  const float d0 = v0 - mean, d1 = v1 - mean, d2 = v2 - mean;
  red[t] = d0 * d0 + d1 * d1 + d2 * d2;
  __syncthreads();
  for (int off = 128; off > 0; off >>= 1) {
    if (t < off) red[t] += red[t + off];
    __syncthreads();
  }
  const float rinv = rsqrtf(red[0] * (1.0f / 768.0f) + 1e-9f);

  __bf16* h = H + (size_t)m * DIMC;
  h[t]       = f2bf(d0 * rinv * gs[t]       + gb[t]);
  h[t + 256] = f2bf(d1 * rinv * gs[t + 256] + gb[t + 256]);
  h[t + 512] = f2bf(d2 * rinv * gs[t + 512] + gb[t + 512]);
}

// =====================================================================
// Attention: one workgroup per (batch, head). 49 tokens padded to 64.
// S = (Q/8) K^T via WMMA -> LDS softmax (mask k>=49) -> O = P V via WMMA.
// Padded query rows are exactly zero -> no NaN; padded keys get P = 0.
// =====================================================================
__global__ __launch_bounds__(128) void attention(
    const __bf16* __restrict__ QKV, __bf16* __restrict__ O)
{
  __shared__ __bf16 sQ[64][64];   // Q, later reused for P (probs)
  __shared__ __bf16 sK[64][64];   // [token][d] == transposed-B layout for S
  __shared__ __bf16 sVT[64][64];  // [d][token] == transposed-B layout for O
  __shared__ float  sS[64][65];

  const int bh = blockIdx.x;
  const int b  = bh / NHEAD;
  const int h  = bh % NHEAD;
  const int tid  = threadIdx.x;
  const int lane = tid & 31;
  const int wave = tid >> 5;

  for (int e = tid; e < 64 * 64; e += 128) {
    const int t = e >> 6, d = e & 63;
    __bf16 q = (__bf16)0.0f, k = (__bf16)0.0f, v = (__bf16)0.0f;
    if (t < KEEPT) {
      const size_t row = (size_t)(b * KEEPT + t) * 2304 + h * 64 + d;
      q = f2bf(bf2f(QKV[row]) * 0.125f);   // fold 1/sqrt(64)
      k = QKV[row + 768];
      v = QKV[row + 1536];
    }
    sQ[t][d] = q; sK[t][d] = k; sVT[d][t] = v;
  }
  __syncthreads();

  const int ln  = lane & 15;
  const int h8  = (lane >> 4) * 8;
  const int h16 = (lane >> 4) * 16;
  const int hi8 = (lane >> 4) * 8;

  const v8f zero = {0.f,0.f,0.f,0.f,0.f,0.f,0.f,0.f};
  v8f acc[4];
  acc[0] = zero; acc[1] = zero; acc[2] = zero; acc[3] = zero;

  // S = Q K^T : wave handles 16 query rows x 64 key cols, K-dim = 64
#pragma unroll
  for (int kk = 0; kk < 64; kk += 32) {
    v16bf a;
    const int m = wave * 16 + ln;
#pragma unroll
    for (int e = 0; e < 8; ++e) {
      a[e]     = sQ[m][kk + h16 + e];
      a[8 + e] = sQ[m][kk + h16 + 8 + e];
    }
#pragma unroll
    for (int j = 0; j < 4; ++j) {
      v16bf bb;
      const int n = j * 16 + ln;
#pragma unroll
      for (int e = 0; e < 16; ++e) bb[e] = sK[n][kk + h16 + e];
      acc[j] = __builtin_amdgcn_wmma_f32_16x16x32_bf16(
          false, a, false, bb, (short)0, acc[j], false, false);
    }
  }
#pragma unroll
  for (int j = 0; j < 4; ++j)
#pragma unroll
    for (int r = 0; r < 8; ++r)
      sS[wave * 16 + r + hi8][j * 16 + ln] = acc[j][r];
  __syncthreads();

  // row softmax with key mask (only first 49 keys valid)
  if (tid < 64) {
    float mx = -1e30f;
    for (int n = 0; n < KEEPT; ++n) mx = fmaxf(mx, sS[tid][n]);
    float ssum = 0.0f;
    for (int n = 0; n < KEEPT; ++n) {
      const float ev = __expf(sS[tid][n] - mx);
      ssum += ev;
      sS[tid][n] = ev;
    }
    const float inv = __builtin_amdgcn_rcpf(ssum);
    for (int n = 0; n < 64; ++n)
      sQ[tid][n] = (n < KEEPT) ? f2bf(sS[tid][n] * inv) : (__bf16)0.0f;
  }
  __syncthreads();

  // O = P V : K-dim is token index (64, padded keys have P==0)
  v8f oc[4];
  oc[0] = zero; oc[1] = zero; oc[2] = zero; oc[3] = zero;
#pragma unroll
  for (int kk = 0; kk < 64; kk += 32) {
    v16bf a;
    const int m = wave * 16 + ln;
#pragma unroll
    for (int e = 0; e < 8; ++e) {
      a[e]     = sQ[m][kk + h16 + e];
      a[8 + e] = sQ[m][kk + h16 + 8 + e];
    }
#pragma unroll
    for (int j = 0; j < 4; ++j) {
      v16bf bb;
      const int n = j * 16 + ln;   // n = head-dim index
#pragma unroll
      for (int e = 0; e < 16; ++e) bb[e] = sVT[n][kk + h16 + e];
      oc[j] = __builtin_amdgcn_wmma_f32_16x16x32_bf16(
          false, a, false, bb, (short)0, oc[j], false, false);
    }
  }
#pragma unroll
  for (int j = 0; j < 4; ++j)
#pragma unroll
    for (int r = 0; r < 8; ++r) {
      const int t = wave * 16 + r + hi8;
      if (t < KEEPT)
        O[(size_t)(b * KEEPT + t) * DIMC + h * 64 + j * 16 + ln] = f2bf(oc[j][r]);
    }
}

// =====================================================================
extern "C" void kernel_launch(void* const* d_in, const int* in_sizes, int n_in,
                              void* d_out, int out_size, void* d_ws, size_t ws_size,
                              hipStream_t stream)
{
  const float* inputs = (const float*)d_in[0];
  const float* conv_w = (const float*)d_in[1];
  /* d_in[2] = ct : class token never selected by mask_idx (all indices >= 1) */
  const float* pe     = (const float*)d_in[3];
  const float* ln1_s  = (const float*)d_in[4];
  const float* ln1_b  = (const float*)d_in[5];
  const float* qkv_w  = (const float*)d_in[6];
  const float* proj_w = (const float*)d_in[7];
  const float* ln2_s  = (const float*)d_in[8];
  const float* ln2_b  = (const float*)d_in[9];
  const float* w1     = (const float*)d_in[10];
  const float* b1     = (const float*)d_in[11];
  const float* w2     = (const float*)d_in[12];
  const float* b2     = (const float*)d_in[13];
  const int*   midx   = (const int*)d_in[14];
  (void)in_sizes; (void)n_in; (void)out_size; (void)ws_size;

  char* ws = (char*)d_ws;
  size_t off = 0;
  auto wsalloc = [&](size_t bytes) -> void* {
    void* p = ws + off;
    off += (bytes + 255) & ~(size_t)255;
    return p;
  };
  float*  X   = (float*) wsalloc((size_t)MTOK * DIMC * 4);   // fp32 residual
  __bf16* Hb  = (__bf16*)wsalloc((size_t)MTOK * DIMC * 2);   // LN output
  __bf16* QKV = (__bf16*)wsalloc((size_t)MTOK * 2304 * 2);
  __bf16* Ob  = (__bf16*)wsalloc((size_t)MTOK * DIMC * 2);   // attn output
  __bf16* Ub  = (__bf16*)wsalloc((size_t)MTOK * FFD  * 2);   // ffn hidden

  const dim3 blk(128);
  patch_embed<<<dim3(DIMC / 128, MTOK / 64), blk, 0, stream>>>(inputs, conv_w, pe, midx, X);

  for (int l = 0; l < 12; ++l) {
    layernorm<<<MTOK, 256, 0, stream>>>(X, ln1_s + l * DIMC, ln1_b + l * DIMC, Hb);
    // qkv: plain bf16 store
    gemm_wmma<false,false,false,true><<<dim3(2304 / 128, MTOK / 64), blk, 0, stream>>>(
        Hb, qkv_w + (size_t)l * DIMC * 2304, nullptr, nullptr, QKV, 2304, DIMC);
    attention<<<64 * NHEAD, blk, 0, stream>>>(QKV, Ob);
    // proj: fp32 residual accumulate into X
    gemm_wmma<false,false,true,false><<<dim3(DIMC / 128, MTOK / 64), blk, 0, stream>>>(
        Ob, proj_w + (size_t)l * DIMC * DIMC, nullptr, X, nullptr, DIMC, DIMC);
    layernorm<<<MTOK, 256, 0, stream>>>(X, ln2_s + l * DIMC, ln2_b + l * DIMC, Hb);
    // ffn1: bias + GELU, bf16 store
    gemm_wmma<true,true,false,true><<<dim3(FFD / 128, MTOK / 64), blk, 0, stream>>>(
        Hb, w1 + (size_t)l * DIMC * FFD, b1 + (size_t)l * FFD, nullptr, Ub, FFD, DIMC);
    // ffn2: bias + fp32 residual accumulate into X
    gemm_wmma<true,false,true,false><<<dim3(DIMC / 128, MTOK / 64), blk, 0, stream>>>(
        Ub, w2 + (size_t)l * FFD * DIMC, b2 + (size_t)l * DIMC, X, nullptr, DIMC, FFD);
  }

  (void)hipMemcpyAsync(d_out, X, (size_t)MTOK * DIMC * 4, hipMemcpyDeviceToDevice, stream);
}
